// MNN4Rec_update_6116033429566
// MI455X (gfx1250) — matrix operand order repaired and checked by reading, not compiled, not run on hardware
//
#include <hip/hip_runtime.h>
#include <math.h>

// ---------------------------------------------------------------------------
// CDNA5 (gfx1250) WMMA helpers: D = A(16x32 bf16) x B(32x16 bf16) + C(f32)
// ---------------------------------------------------------------------------
typedef __attribute__((ext_vector_type(16))) __bf16 bf16x16;
typedef __attribute__((ext_vector_type(8)))  float  f32x8;

union FragAB { bf16x16 v; uint4 q[2]; };

__device__ __forceinline__ f32x8 fzero8() {
  f32x8 z = {0.f,0.f,0.f,0.f,0.f,0.f,0.f,0.f};
  return z;
}

__device__ __forceinline__ f32x8 wmma_bf16(bf16x16 a, bf16x16 b, f32x8 c) {
  // (neg_a, A, neg_b, B, c_mod, C, reuse_a, reuse_b)
  return __builtin_amdgcn_wmma_f32_16x16x32_bf16(false, a, false, b, (short)0, c, false, false);
}

// A fragment from LDS, row-major [rows][ld] bf16.
// ISA layout: lane m = lane&15; k-chunks (lane>>4)*8 .. +7 and +16 .. +23.
__device__ __forceinline__ bf16x16 load_a(const __bf16* base, int ld, int mtile,
                                          int kstep, int lane) {
  int m  = mtile * 16 + (lane & 15);
  int kb = kstep * 32 + ((lane >> 4) << 3);
  const __bf16* p = base + m * ld + kb;
  FragAB f;
  f.q[0] = *(const uint4*)(p);
  f.q[1] = *(const uint4*)(p + 16);
  return f.v;
}

// B fragment from pre-packed weights: one 32B contiguous chunk per lane.
__device__ __forceinline__ bf16x16 load_b(const __bf16* packed, int ntile, int KS,
                                          int kstep, int lane) {
  FragAB f;
  f.v = ((const bf16x16*)packed)[(size_t)(ntile * KS + kstep) * 32 + lane];
  return f.v;
}

// ---------------------------------------------------------------------------
// Pack fp32 weight [K,N] into WMMA B-fragment order (zero-padded).
// Fragment (ntile,kstep): lane holds n = ntile*16+(lane&15),
// k = kstep*32 + (lane>>4)*8 + j (j<8), +16 + (j-8) (j>=8) (bf16-paired).
// ---------------------------------------------------------------------------
__global__ void pack_w(const float* __restrict__ W, __bf16* __restrict__ dst,
                       int K, int N, int KS) {
  int tile = blockIdx.x;               // = ntile*KS + kstep
  int nt = tile / KS, ks = tile % KS;
  int lane = threadIdx.x;
  int n  = nt * 16 + (lane & 15);
  int kb = ks * 32 + ((lane >> 4) << 3);
  FragAB f;
#pragma unroll
  for (int j = 0; j < 16; ++j) {
    int k = kb + j + ((j >> 3) << 3);
    float v = (k < K && n < N) ? W[(size_t)k * N + n] : 0.f;
    f.v[j] = (__bf16)v;
  }
  ((bf16x16*)dst)[(size_t)tile * 32 + lane] = f.v;
}

// ---------------------------------------------------------------------------
// Word branch: x[b]:[30,300] -> MHA (H=20,AD=20) -> LN -> additive -> tanh
// One news item per 256-thread block (8 wave32).  LDS < 60 KB.
// ---------------------------------------------------------------------------
#define LDK_X 320   // WD padded
#define KS_X  10

__global__ __launch_bounds__(256)
void word_kernel(const float* __restrict__ x,
                 const __bf16* __restrict__ wq_p, const __bf16* __restrict__ wk_p,
                 const __bf16* __restrict__ wv_p,
                 const float* __restrict__ bq, const float* __restrict__ bk,
                 const float* __restrict__ bv,
                 const float* __restrict__ ln_g, const float* __restrict__ ln_b,
                 const __bf16* __restrict__ wa_p, const float* __restrict__ wa_b,
                 const float* __restrict__ wa_q,
                 __bf16* __restrict__ wrep) {
  __shared__ __align__(16) unsigned char smem[59136];
  __bf16* xbf   = (__bf16*)(smem);            // [32][320]  20480 B
  __bf16* qs    = (__bf16*)(smem + 20480);    // [30][80]    4800 B
  __bf16* ksm   = (__bf16*)(smem + 25280);    // [30][80]
  __bf16* vsm   = (__bf16*)(smem + 30080);    // [30][80]
  __bf16* outb  = (__bf16*)(smem + 34880);    // [30][400]  24000 B
  __bf16* outbf = (__bf16*)(smem);            // [32][416] overlay (26624 B)
  float*  sco   = (float*)(smem + 58880);     // [32]
  float*  alp   = (float*)(smem + 59008);     // [32]

  const int b = blockIdx.x;
  const int tid = threadIdx.x;
  const int lane = tid & 31;
  const int wid  = tid >> 5;
  const int hi   = lane >> 4;

  // Stage x[b] as bf16, padded to 32x320.
  const float* xb = x + (size_t)b * 30 * 300;
  for (int i = tid; i < 32 * 320; i += 256) {
    int r = i / 320, c = i % 320;
    xbf[i] = (__bf16)((r < 30 && c < 300) ? xb[r * 300 + c] : 0.f);
  }
  __syncthreads();

  // 5 head-groups of 4 heads (80 cols = 5 N-tiles each).
  for (int hg = 0; hg < 5; ++hg) {
    // 15 tile-columns = {Q,K,V} x 5 ntiles, distributed over 8 waves.
    for (int tc = wid; tc < 15; tc += 8) {
      int mat = tc / 5, ntl = tc % 5;
      const __bf16* wp   = (mat == 0) ? wq_p : ((mat == 1) ? wk_p : wv_p);
      const float*  bias = (mat == 0) ? bq   : ((mat == 1) ? bk   : bv);
      __bf16*       dst  = (mat == 0) ? qs   : ((mat == 1) ? ksm  : vsm);
      int ntile = hg * 5 + ntl;       // global N-tile 0..24
      f32x8 c0 = fzero8(), c1 = fzero8();
      for (int ksp = 0; ksp < KS_X; ++ksp) {
        bf16x16 fb = load_b(wp, ntile, KS_X, ksp, lane);
        c0 = wmma_bf16(load_a(xbf, LDK_X, 0, ksp, lane), fb, c0);
        c1 = wmma_bf16(load_a(xbf, LDK_X, 1, ksp, lane), fb, c1);
      }
      int nloc = ntl * 16 + (lane & 15);       // 0..79 inside group
      float bia = bias[hg * 80 + nloc];
#pragma unroll
      for (int r = 0; r < 8; ++r) {
        int m0 = r + 8 * hi;
        if (m0 < 30)      dst[m0 * 80 + nloc]        = (__bf16)(c0[r] + bia);
        if (m0 + 16 < 30) dst[(m0 + 16) * 80 + nloc] = (__bf16)(c1[r] + bia);
      }
    }
    __syncthreads();

    // Attention for this group's 4 heads: 30 rows x 4 heads = 120 jobs.
    for (int p = tid; p < 120; p += 256) {
      int m = p % 30, hl = p / 30;
      const __bf16* qrow = qs + m * 80 + hl * 20;
      // pass 1: row max
      float mx = -1e30f;
      for (int j = 0; j < 30; ++j) {
        const __bf16* krow = ksm + j * 80 + hl * 20;
        float acc = 0.f;
#pragma unroll
        for (int d = 0; d < 20; ++d) acc += (float)qrow[d] * (float)krow[d];
        mx = fmaxf(mx, acc * 0.22360679775f);
      }
      // pass 2: exp-weighted sum of V
      float o[20];
#pragma unroll
      for (int d = 0; d < 20; ++d) o[d] = 0.f;
      float sum = 0.f;
      for (int j = 0; j < 30; ++j) {
        const __bf16* krow = ksm + j * 80 + hl * 20;
        const __bf16* vrow = vsm + j * 80 + hl * 20;
        float acc = 0.f;
#pragma unroll
        for (int d = 0; d < 20; ++d) acc += (float)qrow[d] * (float)krow[d];
        float e = __expf(acc * 0.22360679775f - mx);
        sum += e;
#pragma unroll
        for (int d = 0; d < 20; ++d) o[d] += e * (float)vrow[d];
      }
      float inv = 1.f / sum;
      __bf16* orow = outb + m * 400 + hg * 80 + hl * 20;
#pragma unroll
      for (int d = 0; d < 20; ++d) orow[d] = (__bf16)(o[d] * inv);
    }
    __syncthreads();
  }

  // outbf overlay pads: rows 30,31 full; cols 400..415 on rows 0..29.
  for (int i = tid; i < 2 * 416; i += 256)
    outbf[(30 + i / 416) * 416 + (i % 416)] = (__bf16)0.f;
  for (int i = tid; i < 30 * 16; i += 256)
    outbf[(i / 16) * 416 + 400 + (i % 16)] = (__bf16)0.f;
  if (tid < 32) sco[tid] = 0.f;

  // LayerNorm per row (wave per row, lane-strided 400, shfl reduce).
  for (int m = wid; m < 30; m += 8) {
    float sum = 0.f, sq = 0.f;
    for (int c = lane; c < 400; c += 32) {
      float v = (float)outb[m * 400 + c];
      sum += v; sq += v * v;
    }
#pragma unroll
    for (int off = 16; off > 0; off >>= 1) {
      sum += __shfl_xor(sum, off, 32);
      sq  += __shfl_xor(sq,  off, 32);
    }
    float mu   = sum * (1.f / 400.f);
    float var  = sq * (1.f / 400.f) - mu * mu;
    float rstd = rsqrtf(var + 1e-5f);
    for (int c = lane; c < 400; c += 32) {
      float v = ((float)outb[m * 400 + c] - mu) * rstd * ln_g[c] + ln_b[c];
      outbf[m * 416 + c] = (__bf16)v;
    }
  }
  __syncthreads();

  // Additive attention projection: [32,416] x wa_w[416p,208p] -> tanh -> dot q.
  for (int tc = wid; tc < 13; tc += 8) {
    f32x8 c0 = fzero8(), c1 = fzero8();
    for (int ksp = 0; ksp < 13; ++ksp) {
      bf16x16 fb = load_b(wa_p, tc, 13, ksp, lane);
      c0 = wmma_bf16(load_a(outbf, 416, 0, ksp, lane), fb, c0);
      c1 = wmma_bf16(load_a(outbf, 416, 1, ksp, lane), fb, c1);
    }
    int n = tc * 16 + (lane & 15);
    if (n < 200) {
      float wb = wa_b[n], wq_ = wa_q[n];
#pragma unroll
      for (int r = 0; r < 8; ++r) {
        int m0 = r + 8 * hi;
        if (m0 < 30)      atomicAdd(&sco[m0],      tanhf(c0[r] + wb) * wq_);
        if (m0 + 16 < 30) atomicAdd(&sco[m0 + 16], tanhf(c1[r] + wb) * wq_);
      }
    }
  }
  __syncthreads();
  if (tid == 0) {
    float mx = -1e30f;
    for (int m = 0; m < 30; ++m) mx = fmaxf(mx, sco[m]);
    float s = 0.f;
    for (int m = 0; m < 30; ++m) { float e = __expf(sco[m] - mx); alp[m] = e; s += e; }
    float inv = 1.f / s;
    for (int m = 0; m < 30; ++m) alp[m] *= inv;
  }
  __syncthreads();
  __bf16* wout = wrep + (size_t)b * 400;
  for (int n = tid; n < 400; n += 256) {
    float acc = 0.f;
    for (int m = 0; m < 30; ++m) acc += alp[m] * (float)outbf[m * 416 + n];
    wout[n] = (__bf16)tanhf(acc);
  }
}

// ---------------------------------------------------------------------------
// Entity branch: fc3 -> GCN -> gcn_w -> LN -> additive -> tanh.  1 news/block.
// ---------------------------------------------------------------------------
__global__ __launch_bounds__(256)
void entity_kernel(const float* __restrict__ ent,
                   const __bf16* __restrict__ fc3_p, const float* __restrict__ fc3_b,
                   const float* __restrict__ gA,
                   const __bf16* __restrict__ gcn_p, const float* __restrict__ gcn_b,
                   const float* __restrict__ ln_g, const float* __restrict__ ln_b,
                   const __bf16* __restrict__ ea_p, const float* __restrict__ ea_b,
                   const float* __restrict__ ea_q,
                   __bf16* __restrict__ erep) {
  __shared__ __align__(16) unsigned char smem[33664];
  __bf16* ebf = (__bf16*)(smem);            // [16][128]
  float*  e1  = (float*)(smem + 4096);      // [10][100]
  __bf16* gbf = (__bf16*)(smem + 8096);     // [16][128]
  __bf16* e2r = (__bf16*)(smem + 12192);    // [10][400]
  __bf16* e2l = (__bf16*)(smem + 20192);    // [16][416]
  float*  sco = (float*)(smem + 33504);     // [16]
  float*  alp = (float*)(smem + 33568);     // [16]

  const int b = blockIdx.x;
  const int tid = threadIdx.x;
  const int lane = tid & 31;
  const int wid  = tid >> 5;
  const int hi   = lane >> 4;

  const float* eb = ent + (size_t)b * 10 * 100;
  for (int i = tid; i < 16 * 128; i += 256) {
    int r = i >> 7, c = i & 127;
    ebf[i] = (__bf16)((r < 10 && c < 100) ? eb[r * 100 + c] : 0.f);
    gbf[i] = (__bf16)0.f;
  }
  for (int i = tid; i < 16 * 416; i += 256) e2l[i] = (__bf16)0.f;
  if (tid < 16) sco[tid] = 0.f;
  __syncthreads();

  // e1 = tanh(ent @ fc3_w + b): 7 N-tiles, K=128 (4 ksteps).
  for (int tc = wid; tc < 7; tc += 8) {
    f32x8 c0 = fzero8();
    for (int ksp = 0; ksp < 4; ++ksp)
      c0 = wmma_bf16(load_a(ebf, 128, 0, ksp, lane), load_b(fc3_p, tc, 4, ksp, lane), c0);
    int n = tc * 16 + (lane & 15);
    if (n < 100) {
      float bi = fc3_b[n];
#pragma unroll
      for (int r = 0; r < 8; ++r) {
        int m = r + 8 * hi;
        if (m < 10) e1[m * 100 + n] = tanhf(c0[r] + bi);
      }
    }
  }
  __syncthreads();

  // g = gcn_A @ e1  (10x10 x 10x100, tiny -> VALU), bf16-packed.
  for (int i = tid; i < 10 * 100; i += 256) {
    int m = i / 100, n = i % 100;
    float acc = 0.f;
#pragma unroll
    for (int f = 0; f < 10; ++f) acc += gA[m * 10 + f] * e1[f * 100 + n];
    gbf[m * 128 + n] = (__bf16)acc;
  }
  __syncthreads();

  // e2 = relu(g @ gcn_w + b): 25 N-tiles.
  for (int tc = wid; tc < 25; tc += 8) {
    f32x8 c0 = fzero8();
    for (int ksp = 0; ksp < 4; ++ksp)
      c0 = wmma_bf16(load_a(gbf, 128, 0, ksp, lane), load_b(gcn_p, tc, 4, ksp, lane), c0);
    int n = tc * 16 + (lane & 15);
    float bi = gcn_b[n];
#pragma unroll
    for (int r = 0; r < 8; ++r) {
      int m = r + 8 * hi;
      if (m < 10) e2r[m * 400 + n] = (__bf16)fmaxf(c0[r] + bi, 0.f);
    }
  }
  __syncthreads();

  // LayerNorm rows 0..9.
  for (int m = wid; m < 10; m += 8) {
    float sum = 0.f, sq = 0.f;
    for (int c = lane; c < 400; c += 32) {
      float v = (float)e2r[m * 400 + c];
      sum += v; sq += v * v;
    }
#pragma unroll
    for (int off = 16; off > 0; off >>= 1) {
      sum += __shfl_xor(sum, off, 32);
      sq  += __shfl_xor(sq,  off, 32);
    }
    float mu = sum * 0.0025f;
    float rstd = rsqrtf(sq * 0.0025f - mu * mu + 1e-5f);
    for (int c = lane; c < 400; c += 32) {
      float v = ((float)e2r[m * 400 + c] - mu) * rstd * ln_g[c] + ln_b[c];
      e2l[m * 416 + c] = (__bf16)v;
    }
  }
  __syncthreads();

  // Additive attention over entities.
  for (int tc = wid; tc < 13; tc += 8) {
    f32x8 c0 = fzero8();
    for (int ksp = 0; ksp < 13; ++ksp)
      c0 = wmma_bf16(load_a(e2l, 416, 0, ksp, lane), load_b(ea_p, tc, 13, ksp, lane), c0);
    int n = tc * 16 + (lane & 15);
    if (n < 200) {
      float ebi = ea_b[n], eq = ea_q[n];
#pragma unroll
      for (int r = 0; r < 8; ++r) {
        int m = r + 8 * hi;
        if (m < 10) atomicAdd(&sco[m], tanhf(c0[r] + ebi) * eq);
      }
    }
  }
  __syncthreads();
  if (tid == 0) {
    float mx = -1e30f;
    for (int m = 0; m < 10; ++m) mx = fmaxf(mx, sco[m]);
    float s = 0.f;
    for (int m = 0; m < 10; ++m) { float e = __expf(sco[m] - mx); alp[m] = e; s += e; }
    float inv = 1.f / s;
    for (int m = 0; m < 10; ++m) alp[m] *= inv;
  }
  __syncthreads();
  __bf16* eo = erep + (size_t)b * 400;
  for (int n = tid; n < 400; n += 256) {
    float acc = 0.f;
    for (int m = 0; m < 10; ++m) acc += alp[m] * (float)e2l[m * 416 + n];
    eo[n] = (__bf16)tanhf(acc);
  }
}

// ---------------------------------------------------------------------------
// Category / subcategory: gather + tanh(emb @ fc + b).  64 news rows / block.
// ---------------------------------------------------------------------------
__global__ __launch_bounds__(256)
void catsub_kernel(const int* __restrict__ cidx, const int* __restrict__ sidx,
                   const float* __restrict__ emb_cat, const float* __restrict__ emb_sub,
                   const __bf16* __restrict__ fc1_p, const float* __restrict__ fc1_b,
                   const __bf16* __restrict__ fc2_p, const float* __restrict__ fc2_b,
                   __bf16* __restrict__ crep, __bf16* __restrict__ srep) {
  __shared__ __align__(16) __bf16 abf[64 * 64];
  const int blk = blockIdx.x;
  const int tid = threadIdx.x;
  const int lane = tid & 31;
  const int wid  = tid >> 5;
  const int hi   = lane >> 4;

  for (int pass = 0; pass < 2; ++pass) {
    const int*   idx = pass ? sidx : cidx;
    const float* emb = pass ? emb_sub : emb_cat;
    const __bf16* wp = pass ? fc2_p : fc1_p;
    const float* bb  = pass ? fc2_b : fc1_b;
    __bf16*      rep = pass ? srep : crep;

    for (int i = tid; i < 64 * 64; i += 256) {
      int r = i >> 6, c = i & 63;
      int bg = blk * 64 + r;
      abf[i] = (__bf16)((c < 50) ? emb[(size_t)idx[bg] * 50 + c] : 0.f);
    }
    __syncthreads();

    for (int tc = wid; tc < 25; tc += 8) {
      f32x8 acc[4];
#pragma unroll
      for (int mt = 0; mt < 4; ++mt) acc[mt] = fzero8();
      for (int ksp = 0; ksp < 2; ++ksp) {
        bf16x16 fb = load_b(wp, tc, 2, ksp, lane);
#pragma unroll
        for (int mt = 0; mt < 4; ++mt)
          acc[mt] = wmma_bf16(load_a(abf, 64, mt, ksp, lane), fb, acc[mt]);
      }
      int n = tc * 16 + (lane & 15);
      float bi = bb[n];
#pragma unroll
      for (int mt = 0; mt < 4; ++mt)
#pragma unroll
        for (int r = 0; r < 8; ++r) {
          int m = mt * 16 + r + 8 * hi;
          rep[(size_t)(blk * 64 + m) * 400 + n] = (__bf16)tanhf(acc[mt][r] + bi);
        }
    }
    __syncthreads();
  }
}

// ---------------------------------------------------------------------------
// Final fusion: additive attention over 4 views, then tanh.  16 news / block.
// ---------------------------------------------------------------------------
__global__ __launch_bounds__(256)
void fuse_kernel(const __bf16* __restrict__ wrep, const __bf16* __restrict__ erep,
                 const __bf16* __restrict__ crep, const __bf16* __restrict__ srep,
                 const __bf16* __restrict__ na_p, const float* __restrict__ na_b,
                 const float* __restrict__ na_q,
                 float* __restrict__ out) {
  __shared__ __align__(16) __bf16 vbf[64 * 416];
  __shared__ float sco[64];
  __shared__ float alp[64];
  const int blk = blockIdx.x;
  const int tid = threadIdx.x;
  const int lane = tid & 31;
  const int wid  = tid >> 5;
  const int hi   = lane >> 4;

  const __bf16* views[4] = {wrep, erep, crep, srep};
  for (int i = tid; i < 64 * 416; i += 256) {
    int r = i / 416, c = i % 416;
    int nl = r >> 2, v = r & 3;
    __bf16 val = (__bf16)0.f;
    if (c < 400) val = views[v][(size_t)(blk * 16 + nl) * 400 + c];
    vbf[i] = val;
  }
  if (tid < 64) sco[tid] = 0.f;
  __syncthreads();

  for (int tc = wid; tc < 13; tc += 8) {
    f32x8 acc[4];
#pragma unroll
    for (int mt = 0; mt < 4; ++mt) acc[mt] = fzero8();
    for (int ksp = 0; ksp < 13; ++ksp) {
      bf16x16 fb = load_b(na_p, tc, 13, ksp, lane);
#pragma unroll
      for (int mt = 0; mt < 4; ++mt)
        acc[mt] = wmma_bf16(load_a(vbf, 416, mt, ksp, lane), fb, acc[mt]);
    }
    int n = tc * 16 + (lane & 15);
    if (n < 200) {
      float nb = na_b[n], nq = na_q[n];
#pragma unroll
      for (int mt = 0; mt < 4; ++mt)
#pragma unroll
        for (int r = 0; r < 8; ++r) {
          int m = mt * 16 + r + 8 * hi;
          atomicAdd(&sco[m], tanhf(acc[mt][r] + nb) * nq);
        }
    }
  }
  __syncthreads();
  if (tid < 16) {
    float s0 = sco[tid * 4], s1 = sco[tid * 4 + 1];
    float s2 = sco[tid * 4 + 2], s3 = sco[tid * 4 + 3];
    float mx = fmaxf(fmaxf(s0, s1), fmaxf(s2, s3));
    float e0 = __expf(s0 - mx), e1 = __expf(s1 - mx);
    float e2 = __expf(s2 - mx), e3 = __expf(s3 - mx);
    float inv = 1.f / (e0 + e1 + e2 + e3);
    alp[tid * 4] = e0 * inv; alp[tid * 4 + 1] = e1 * inv;
    alp[tid * 4 + 2] = e2 * inv; alp[tid * 4 + 3] = e3 * inv;
  }
  __syncthreads();
  for (int i = tid; i < 16 * 400; i += 256) {
    int nl = i / 400, n = i % 400;
    float acc = 0.f;
#pragma unroll
    for (int v = 0; v < 4; ++v)
      acc += alp[nl * 4 + v] * (float)vbf[(nl * 4 + v) * 416 + n];
    out[(size_t)(blk * 16 + nl) * 400 + n] = tanhf(acc);
  }
}

// ---------------------------------------------------------------------------
extern "C" void kernel_launch(void* const* d_in, const int* in_sizes, int n_in,
                              void* d_out, int out_size, void* d_ws, size_t ws_size,
                              hipStream_t stream) {
  const float* x       = (const float*)d_in[0];
  const float* ent     = (const float*)d_in[1];
  const int*   cidx    = (const int*)d_in[2];
  const int*   sidx    = (const int*)d_in[3];
  const float* emb_cat = (const float*)d_in[4];
  const float* fc1_w   = (const float*)d_in[5];
  const float* fc1_b   = (const float*)d_in[6];
  const float* emb_sub = (const float*)d_in[7];
  const float* fc2_w   = (const float*)d_in[8];
  const float* fc2_b   = (const float*)d_in[9];
  const float* wq      = (const float*)d_in[10];
  const float* bq      = (const float*)d_in[11];
  const float* wk      = (const float*)d_in[12];
  const float* bk      = (const float*)d_in[13];
  const float* wv      = (const float*)d_in[14];
  const float* bv      = (const float*)d_in[15];
  const float* ln_g    = (const float*)d_in[16];
  const float* ln_b    = (const float*)d_in[17];
  const float* wa_w    = (const float*)d_in[18];
  const float* wa_b    = (const float*)d_in[19];
  const float* wa_q    = (const float*)d_in[20];
  const float* fc3_w   = (const float*)d_in[21];
  const float* fc3_b   = (const float*)d_in[22];
  const float* gcn_A   = (const float*)d_in[23];
  const float* gcn_w   = (const float*)d_in[24];
  const float* gcn_b   = (const float*)d_in[25];
  const float* ea_w    = (const float*)d_in[26];
  const float* ea_b    = (const float*)d_in[27];
  const float* ea_q    = (const float*)d_in[28];
  const float* na_w    = (const float*)d_in[29];
  const float* na_b    = (const float*)d_in[30];
  const float* na_q    = (const float*)d_in[31];
  float* out = (float*)d_out;
  (void)in_sizes; (void)n_in; (void)out_size; (void)ws_size;

  unsigned char* ws = (unsigned char*)d_ws;
  size_t off = 0;
  auto alloc = [&](size_t bytes) -> void* {
    void* p = ws + off;
    off = (off + bytes + 255) & ~(size_t)255;
    return p;
  };
  // Packed bf16 weights: NT*KS fragments of 1024 B each.
  __bf16* wq_p  = (__bf16*)alloc(25 * 10 * 1024);
  __bf16* wk_p  = (__bf16*)alloc(25 * 10 * 1024);
  __bf16* wv_p  = (__bf16*)alloc(25 * 10 * 1024);
  __bf16* wa_p  = (__bf16*)alloc(13 * 13 * 1024);
  __bf16* ea_p  = (__bf16*)alloc(13 * 13 * 1024);
  __bf16* na_p  = (__bf16*)alloc(13 * 13 * 1024);
  __bf16* fc3_p = (__bf16*)alloc(7  * 4  * 1024);
  __bf16* gcn_p = (__bf16*)alloc(25 * 4  * 1024);
  __bf16* fc1_p = (__bf16*)alloc(25 * 2  * 1024);
  __bf16* fc2_p = (__bf16*)alloc(25 * 2  * 1024);
  // View buffers (bf16), [4096, 400].
  __bf16* wrep = (__bf16*)alloc((size_t)4096 * 400 * 2);
  __bf16* erep = (__bf16*)alloc((size_t)4096 * 400 * 2);
  __bf16* crep = (__bf16*)alloc((size_t)4096 * 400 * 2);
  __bf16* srep = (__bf16*)alloc((size_t)4096 * 400 * 2);

  pack_w<<<25 * 10, 32, 0, stream>>>(wq,    wq_p,  300, 400, 10);
  pack_w<<<25 * 10, 32, 0, stream>>>(wk,    wk_p,  300, 400, 10);
  pack_w<<<25 * 10, 32, 0, stream>>>(wv,    wv_p,  300, 400, 10);
  pack_w<<<13 * 13, 32, 0, stream>>>(wa_w,  wa_p,  400, 200, 13);
  pack_w<<<13 * 13, 32, 0, stream>>>(ea_w,  ea_p,  400, 200, 13);
  pack_w<<<13 * 13, 32, 0, stream>>>(na_w,  na_p,  400, 200, 13);
  pack_w<<<7  * 4,  32, 0, stream>>>(fc3_w, fc3_p, 100, 100, 4);
  pack_w<<<25 * 4,  32, 0, stream>>>(gcn_w, gcn_p, 100, 400, 4);
  pack_w<<<25 * 2,  32, 0, stream>>>(fc1_w, fc1_p, 50,  400, 2);
  pack_w<<<25 * 2,  32, 0, stream>>>(fc2_w, fc2_p, 50,  400, 2);

  word_kernel<<<4096, 256, 0, stream>>>(x, wq_p, wk_p, wv_p, bq, bk, bv,
                                        ln_g, ln_b, wa_p, wa_b, wa_q, wrep);
  entity_kernel<<<4096, 256, 0, stream>>>(ent, fc3_p, fc3_b, gcn_A, gcn_p, gcn_b,
                                          ln_g, ln_b, ea_p, ea_b, ea_q, erep);
  catsub_kernel<<<64, 256, 0, stream>>>(cidx, sidx, emb_cat, emb_sub,
                                        fc1_p, fc1_b, fc2_p, fc2_b, crep, srep);
  fuse_kernel<<<256, 256, 0, stream>>>(wrep, erep, crep, srep,
                                       na_p, na_b, na_q, out);
}
